// Sparse_Attention_87359634801115
// MI455X (gfx1250) — matrix operand order repaired and checked by reading, not compiled, not run on hardware
//
#include <hip/hip_runtime.h>
#include <hip/hip_bf16.h>

typedef __attribute__((ext_vector_type(16))) __bf16      v16bf;
typedef __attribute__((ext_vector_type(8)))  float       v8f;
typedef __attribute__((ext_vector_type(8)))  unsigned int v8u;

#define B_   4
#define S_   4096
#define D_   1024
#define GH_  8
#define KD_  64
#define L_   256

// ---------- helpers ----------
__device__ __forceinline__ unsigned short f2bf(float f) {
  unsigned int u = __builtin_bit_cast(unsigned int, f);
  u += 0x7FFFu + ((u >> 16) & 1u);           // round-to-nearest-even
  return (unsigned short)(u >> 16);
}

__device__ __forceinline__ v16bf make_frag(uint4 lo, uint4 hi) {
  v8u t;
  t[0] = lo.x; t[1] = lo.y; t[2] = lo.z; t[3] = lo.w;
  t[4] = hi.x; t[5] = hi.y; t[6] = hi.z; t[7] = hi.w;
  return __builtin_bit_cast(v16bf, t);
}

// CDNA5 async global->LDS copy (ASYNCcnt-tracked DMA, bypasses VGPRs).
// VDST = LDS byte offset (VGPR), VADDR = 64-bit global address (VGPR pair).
__device__ __forceinline__ void async_ld_b128(const void* g, void* lds) {
  unsigned loff = (unsigned)(unsigned long long)lds;   // low 32 bits used as LDS byte offset
  asm volatile("global_load_async_to_lds_b128 %0, %1, off"
               :: "v"(loff), "v"(g)
               : "memory");
}
__device__ __forceinline__ void wait_async0() {
  asm volatile("s_wait_asynccnt 0" ::: "memory");
}

// ---------- fp32 -> bf16 conversion ----------
__global__ void k_f32_to_bf16(const float* __restrict__ src,
                              unsigned short* __restrict__ dst, int n) {
  for (int i = blockIdx.x * blockDim.x + threadIdx.x; i < n;
       i += gridDim.x * blockDim.x)
    dst[i] = f2bf(src[i]);
}

// ---------- bf16 GEMM: Y[M,N] = X[M,K] * W[N,K]^T + bias ----------
// Block: 256 threads (8 waves). Tile 128(M) x 128(N) x 32(K), double-buffered
// LDS filled by async global->LDS DMA; WMMA issues overlap the DMA.
// Wave w: M-strip = (w&3)*32, N-strip = (w>>2)*64 -> 2x4 accumulators of 16x16.
template <bool STORE_BF16>
__global__ __launch_bounds__(256) void k_gemm_bf16(
    const unsigned short* __restrict__ X,   // [M,K] bf16
    const unsigned short* __restrict__ W,   // [N,K] bf16
    const float* __restrict__ bias,         // [N] fp32
    void* __restrict__ out,                 // [M,N] bf16 or fp32
    int M, int N, int K) {
  __shared__ unsigned short xs[2][128][32];
  __shared__ unsigned short wl[2][128][32];

  const int t    = threadIdx.x;
  const int lane = t & 31;
  const int wave = t >> 5;
  const int hlf  = lane >> 4;     // lane half (ISA A/B fragment split)
  const int l16  = lane & 15;
  const int wm   = wave & 3;
  const int wn   = wave >> 2;
  const int m0   = blockIdx.y * 128;
  const int n0   = blockIdx.x * 128;

  // staging: thread t owns 32 contiguous bytes of one row per tile
  const int lr = t >> 1;
  const int lc = (t & 1) << 4;
  const unsigned short* gX = X + (size_t)(m0 + lr) * K + lc;
  const unsigned short* gW = W + (size_t)(n0 + lr) * K + lc;

  v8f acc[2][4] = {};

  // async preload of k-step 0
  async_ld_b128(gX,     &xs[0][lr][lc]);
  async_ld_b128(gX + 8, &xs[0][lr][lc + 8]);
  async_ld_b128(gW,     &wl[0][lr][lc]);
  async_ld_b128(gW + 8, &wl[0][lr][lc + 8]);
  wait_async0();
  __syncthreads();

  const int nk = K >> 5;
  for (int kt = 0; kt < nk; ++kt) {
    const int cur = kt & 1;
    if (kt + 1 < nk) {                       // kick off DMA for next k-step
      const int nxt = cur ^ 1;
      const unsigned short* a = gX + (size_t)(kt + 1) * 32;
      const unsigned short* b = gW + (size_t)(kt + 1) * 32;
      async_ld_b128(a,     &xs[nxt][lr][lc]);
      async_ld_b128(a + 8, &xs[nxt][lr][lc + 8]);
      async_ld_b128(b,     &wl[nxt][lr][lc]);
      async_ld_b128(b + 8, &wl[nxt][lr][lc + 8]);
    }

    // A fragments: lane half 0 -> K {0..15}; half 1 -> K {16..31}
    v16bf af[2];
#pragma unroll
    for (int ms = 0; ms < 2; ++ms) {
      const unsigned short* p = &xs[cur][wm * 32 + ms * 16 + l16][hlf * 16];
      af[ms] = make_frag(*(const uint4*)p, *(const uint4*)(p + 8));
    }
    // B fragments: column N = l16, K = hlf*16 .. hlf*16+15 (contiguous in W row)
    v16bf bfv[4];
#pragma unroll
    for (int ns = 0; ns < 4; ++ns) {
      const unsigned short* p = &wl[cur][wn * 64 + ns * 16 + l16][hlf * 16];
      bfv[ns] = make_frag(*(const uint4*)p, *(const uint4*)(p + 8));
    }
#pragma unroll
    for (int ms = 0; ms < 2; ++ms)
#pragma unroll
      for (int ns = 0; ns < 4; ++ns)
        acc[ms][ns] = __builtin_amdgcn_wmma_f32_16x16x32_bf16(
            false, af[ms], false, bfv[ns], (short)0, acc[ms][ns], false, false);

    wait_async0();      // next buffer's DMA complete
    __syncthreads();    // visible to all waves
  }

  // epilogue: C/D layout -> row M = r + 8*hlf, col N = l16
#pragma unroll
  for (int ms = 0; ms < 2; ++ms)
#pragma unroll
    for (int ns = 0; ns < 4; ++ns) {
      const int gn = n0 + wn * 64 + ns * 16 + l16;
      const float bb = bias[gn];
#pragma unroll
      for (int r = 0; r < 8; ++r) {
        const int gm = m0 + wm * 32 + ms * 16 + r + 8 * hlf;
        const float v = acc[ms][ns][r] + bb;
        if (STORE_BF16)
          ((unsigned short*)out)[(size_t)gm * N + gn] = f2bf(v);
        else
          ((float*)out)[(size_t)gm * N + gn] = v;
      }
    }
}

// ---------- gather k_sel, v_sel^T, mask bias ----------
__global__ void k_gather(const unsigned short* __restrict__ kg,   // [B*S,512]
                         const unsigned short* __restrict__ vbuf, // [B*S,1024]
                         const unsigned char* __restrict__ mask,  // [B*S] bool
                         unsigned short* __restrict__ ksel,       // [bh,256,64]
                         unsigned short* __restrict__ vselT,      // [bh,64,256]
                         float* __restrict__ mbias) {             // [bh,256]
  const int tid = blockIdx.x * blockDim.x + threadIdx.x;          // 2^19 total
  const int d = tid & 63;
  const int l = (tid >> 6) & 255;
  const int h = (tid >> 14) & 7;
  const int b = tid >> 17;
  const int p = (l >> 2) * 64 + 32 + 4 * h + (l & 3);             // gathered pos
  const int bh = b * 8 + h;
  const size_t row = (size_t)(b * S_ + p);
  ksel[((size_t)bh * 256 + l) * 64 + d]  = kg[row * 512 + h * 64 + d];
  vselT[((size_t)bh * 64 + d) * 256 + l] = vbuf[row * 1024 + 512 + h * 64 + d];
  if (d == 0) mbias[bh * 256 + l] = mask[b * S_ + p] ? -1.0e10f : 0.0f;
}

// ---------- attention: QK -> softmax -> AV, writes g_out into vbuf upper ----
__global__ __launch_bounds__(64) void k_attn(
    const unsigned short* __restrict__ qg,     // [B*S,512] bf16
    const unsigned short* __restrict__ ksel,   // [bh,256,64] bf16
    const unsigned short* __restrict__ vselT,  // [bh,64,256] bf16
    const float* __restrict__ mbias,           // [bh,256]
    unsigned short* __restrict__ vbuf) {       // [B*S,1024] bf16
  __shared__ float          sc[2][16][256];    // 32 KB scores
  __shared__ unsigned short pr[2][16][256];    // 16 KB probs

  const int lane = threadIdx.x & 31;
  const int wv   = threadIdx.x >> 5;
  const int hlf  = lane >> 4;
  const int l16  = lane & 15;

  const int sblk = blockIdx.x & 127;           // S/32
  const int bh   = blockIdx.x >> 7;
  const int h    = bh & 7;
  const int b    = bh >> 3;
  const int s0   = sblk * 32 + wv * 16;

  // Q fragments for both K=32 steps (reused across all 16 l-tiles)
  v16bf aq[2];
  const unsigned short* qrow = qg + (size_t)(b * S_ + s0 + l16) * 512 + h * 64;
#pragma unroll
  for (int kt = 0; kt < 2; ++kt) {
    const unsigned short* p = qrow + kt * 32 + hlf * 16;
    aq[kt] = make_frag(*(const uint4*)p, *(const uint4*)(p + 8));
  }

  // QK: 16 tiles of 16 columns
  for (int lt = 0; lt < 16; ++lt) {
    v8f c = {};
    const unsigned short* krow = ksel + ((size_t)bh * 256 + lt * 16 + l16) * 64;
#pragma unroll
    for (int kt = 0; kt < 2; ++kt) {
      const unsigned short* p = krow + kt * 32 + hlf * 16;
      v16bf bfr = make_frag(*(const uint4*)p, *(const uint4*)(p + 8));
      c = __builtin_amdgcn_wmma_f32_16x16x32_bf16(false, aq[kt], false, bfr,
                                                  (short)0, c, false, false);
    }
    const float mb = mbias[bh * 256 + lt * 16 + l16];
#pragma unroll
    for (int r = 0; r < 8; ++r)
      sc[wv][r + 8 * hlf][lt * 16 + l16] = c[r] * 0.125f + mb;   // 1/sqrt(64)
  }

  // softmax over 256 per row (wave-cooperative, 8 cols/lane)
  for (int row = 0; row < 16; ++row) {
    float vals[8], m = -3.4e38f;
#pragma unroll
    for (int j = 0; j < 8; ++j) {
      vals[j] = sc[wv][row][lane * 8 + j];
      m = fmaxf(m, vals[j]);
    }
#pragma unroll
    for (int off = 16; off >= 1; off >>= 1) m = fmaxf(m, __shfl_xor(m, off, 32));
    float s = 0.f;
#pragma unroll
    for (int j = 0; j < 8; ++j) { vals[j] = __expf(vals[j] - m); s += vals[j]; }
#pragma unroll
    for (int off = 16; off >= 1; off >>= 1) s += __shfl_xor(s, off, 32);
    const float inv = 1.0f / s;
#pragma unroll
    for (int j = 0; j < 8; ++j) pr[wv][row][lane * 8 + j] = f2bf(vals[j] * inv);
  }

  // AV: K = 256 -> 8 WMMA steps, 4 output tiles over d
  v8f o[4] = {};
  for (int kt = 0; kt < 8; ++kt) {
    const unsigned short* p = &pr[wv][l16][kt * 32 + hlf * 16];
    v16bf af = make_frag(*(const uint4*)p, *(const uint4*)(p + 8));
#pragma unroll
    for (int nd = 0; nd < 4; ++nd) {
      const unsigned short* vp =
          vselT + ((size_t)bh * 64 + nd * 16 + l16) * 256 + kt * 32 + hlf * 16;
      v16bf bfr = make_frag(*(const uint4*)vp, *(const uint4*)(vp + 8));
      o[nd] = __builtin_amdgcn_wmma_f32_16x16x32_bf16(false, af, false, bfr,
                                                      (short)0, o[nd], false, false);
    }
  }
#pragma unroll
  for (int nd = 0; nd < 4; ++nd)
#pragma unroll
    for (int r = 0; r < 8; ++r) {
      const int row = s0 + r + 8 * hlf;
      vbuf[(size_t)(b * S_ + row) * 1024 + 512 + h * 64 + nd * 16 + l16] =
          f2bf(o[nd][r]);
    }
}

// ---------- host ----------
extern "C" void kernel_launch(void* const* d_in, const int* in_sizes, int n_in,
                              void* d_out, int out_size, void* d_ws,
                              size_t ws_size, hipStream_t stream) {
  (void)in_sizes; (void)n_in; (void)out_size; (void)ws_size;
  const float*         x    = (const float*)d_in[0];
  const unsigned char* mask = (const unsigned char*)d_in[1];  // jnp bool = 1B
  const float* Wq = (const float*)d_in[2];
  const float* bq = (const float*)d_in[3];
  const float* Wk = (const float*)d_in[4];
  const float* bk = (const float*)d_in[5];
  const float* Wv = (const float*)d_in[6];
  const float* bv = (const float*)d_in[7];
  const float* Wo = (const float*)d_in[8];
  const float* bo = (const float*)d_in[9];

  const int M = B_ * S_;  // 16384
  char* ws = (char*)d_ws;
  size_t off = 0;
  auto alloc = [&](size_t bytes) {
    char* p = ws + off;
    off += (bytes + 255) & ~(size_t)255;
    return p;
  };
  unsigned short* xb    = (unsigned short*)alloc((size_t)M * D_ * 2);   // 32 MB
  unsigned short* wqb   = (unsigned short*)alloc((size_t)D_ * D_ * 2);  // 2 MB
  unsigned short* wkb   = (unsigned short*)alloc((size_t)D_ * D_ * 2);
  unsigned short* wvb   = (unsigned short*)alloc((size_t)D_ * D_ * 2);
  unsigned short* wob   = (unsigned short*)alloc((size_t)D_ * D_ * 2);
  unsigned short* qgb   = (unsigned short*)alloc((size_t)M * 512 * 2);  // 16 MB
  unsigned short* kgb   = (unsigned short*)alloc((size_t)M * 512 * 2);  // 16 MB
  unsigned short* vbuf  = (unsigned short*)alloc((size_t)M * D_ * 2);   // 32 MB
  unsigned short* ksel  = (unsigned short*)alloc((size_t)B_ * GH_ * L_ * KD_ * 2);
  unsigned short* vselT = (unsigned short*)alloc((size_t)B_ * GH_ * L_ * KD_ * 2);
  float*          mbias = (float*)alloc((size_t)B_ * GH_ * L_ * 4);

  // fp32 -> bf16
  k_f32_to_bf16<<<2048, 256, 0, stream>>>(x, xb, M * D_);
  k_f32_to_bf16<<<512, 256, 0, stream>>>(Wq, wqb, D_ * D_);
  k_f32_to_bf16<<<512, 256, 0, stream>>>(Wk, wkb, D_ * D_);
  k_f32_to_bf16<<<512, 256, 0, stream>>>(Wv, wvb, D_ * D_);
  k_f32_to_bf16<<<512, 256, 0, stream>>>(Wo, wob, D_ * D_);

  // projections (Q/K only global heads -> upper 512 rows of Wq/Wk)
  k_gemm_bf16<true><<<dim3(512 / 128, M / 128), 256, 0, stream>>>(
      xb, wqb + (size_t)512 * D_, bq + 512, qgb, M, 512, D_);
  k_gemm_bf16<true><<<dim3(512 / 128, M / 128), 256, 0, stream>>>(
      xb, wkb + (size_t)512 * D_, bk + 512, kgb, M, 512, D_);
  k_gemm_bf16<true><<<dim3(1024 / 128, M / 128), 256, 0, stream>>>(
      xb, wvb, bv, vbuf, M, 1024, D_);

  // gather K/V for global heads + mask bias
  k_gather<<<2048, 256, 0, stream>>>(kgb, vbuf, mask, ksel, vselT, mbias);

  // attention: overwrites upper 512 columns of vbuf with g_out -> vbuf == qkv
  k_attn<<<B_ * GH_ * (S_ / 32), 64, 0, stream>>>(qgb, ksel, vselT, mbias, vbuf);

  // output projection -> fp32 d_out
  k_gemm_bf16<false><<<dim3(1024 / 128, M / 128), 256, 0, stream>>>(
      vbuf, wob, bo, (float*)d_out, M, 1024, D_);
}
